// GatedLinearAttention_39951785787599
// MI455X (gfx1250) — compile-verified, matrix operand order checked by reading, not run on hardware
//
#include <hip/hip_runtime.h>

// ---------------------------------------------------------------------------
// Gated linear attention on MI455X (gfx1250, wave32, WMMA f16->f32).
//
// Pipeline (per batch b in [0,16), C=384, N=64*64=4096):
//   K1 groupnorm          : x[C][N] -> XT f16 [N][C]  (transposed via LDS)
//   K2 qkvg GEMM          : Wcat[1920][C] @ XT^T  -> Qt[N][C] (scaled, transposed),
//                           Kb[C][N], Vb[C][N], Ib[C][N] (sigmoid), f-partials
//   K2b fsum reduce       : deterministic reduction of f-gate partials
//   K3 knowledge GEMM     : (Kb*Ib)[C][N] @ Vb^T / N -> knw f32 [C][C]
//   K4 combine            : new_memory = (fsum/N)*prev + knw (f32 out) ; Mt f16 [d][c]
//   K5 retrieved GEMM     : Mt[384][384] @ Qt^T -> Rt f16 [N][C] (transposed, aliases XT)
//   K6 out GEMM           : Wp[384][384] @ Rt^T + bp + x -> out f32
//
// Every GEMM is D = A * B^T with BOTH operands K-contiguous, so A and B
// fragments use the same per-lane load (CDNA5 16-bit A-matrix VGPR layout;
// symmetric layout assumed for B). Wave tile 32x64 = 2x4 WMMA fragments:
// 8 v_wmma per 12 global_load_b128 per K-step.
// ---------------------------------------------------------------------------

typedef __attribute__((ext_vector_type(16))) _Float16 v16h;
typedef __attribute__((ext_vector_type(8)))  _Float16 v8h;
typedef __attribute__((ext_vector_type(8)))  float    v8f;

#define CDIM 384
#define NDIM 4096
#define BDIM 16
#define MROWS 1920          // 5*C rows of the fused projection
#define QSCALE 0.05103103630798287f   // 384^-0.5

__device__ __forceinline__ v8f wmma16(v16h a, v16h b, v8f c) {
  return __builtin_amdgcn_wmma_f32_16x16x32_f16(false, a, false, b, (short)0, c,
                                                false, false);
}

// Fragment load: operand stored row-major [16 rows][K contiguous], ld halves.
__device__ __forceinline__ v16h load_frag(const _Float16* __restrict__ base,
                                          int ld, int lane) {
  int r  = lane & 15;
  int kh = (lane >> 4) << 3;                 // 0 or 8
  const _Float16* p = base + (size_t)r * ld + kh;
  union { v16h v; v8h h[2]; } u;
  u.h[0] = *(const v8h*)(p);
  u.h[1] = *(const v8h*)(p + 16);
  return u.v;
}

// Same, but A = elementwise product of two buffers (k * i_gate), fused in VGPRs.
__device__ __forceinline__ v16h load_frag_mul(const _Float16* __restrict__ kb,
                                              const _Float16* __restrict__ ib,
                                              int ld, int lane) {
  int r  = lane & 15;
  int kh = (lane >> 4) << 3;
  const _Float16* pk = kb + (size_t)r * ld + kh;
  const _Float16* pi = ib + (size_t)r * ld + kh;
  union { v16h v; v8h h[2]; } u;
  u.h[0] = *(const v8h*)(pk)      * *(const v8h*)(pi);
  u.h[1] = *(const v8h*)(pk + 16) * *(const v8h*)(pi + 16);
  return u.v;
}

__device__ __forceinline__ float sigmoidf_(float z) {
  return 1.0f / (1.0f + __expf(-z));
}

// ---------------------------------------------------------------------------
// K0: weight prep — fuse Wq/Wk/Wv/Wg into Wc16 [1920][384] f16, bias concat,
//     Wp -> f16.
// ---------------------------------------------------------------------------
__global__ __launch_bounds__(256) void k_prep(
    const float* __restrict__ Wq, const float* __restrict__ Wk,
    const float* __restrict__ Wv, const float* __restrict__ Wg,
    const float* __restrict__ Wp,
    const float* __restrict__ bq, const float* __restrict__ bk,
    const float* __restrict__ bv, const float* __restrict__ bg,
    _Float16* __restrict__ Wc16, float* __restrict__ bcat,
    _Float16* __restrict__ Wp16) {
  int idx = blockIdx.x * 256 + threadIdx.x;
  if (idx < MROWS * CDIM) {
    int row = idx / CDIM, k = idx % CDIM;
    float v;
    if      (row < 384)  v = Wq[(size_t)row * CDIM + k];
    else if (row < 768)  v = Wk[(size_t)(row - 384) * CDIM + k];
    else if (row < 1152) v = Wv[(size_t)(row - 768) * CDIM + k];
    else                 v = Wg[(size_t)(row - 1152) * CDIM + k];
    Wc16[idx] = (_Float16)v;
  }
  if (idx < CDIM * CDIM) Wp16[idx] = (_Float16)Wp[idx];
  if (idx < MROWS) {
    float v;
    if      (idx < 384)  v = bq[idx];
    else if (idx < 768)  v = bk[idx - 384];
    else if (idx < 1152) v = bv[idx - 768];
    else                 v = bg[idx - 1152];
    bcat[idx] = v;
  }
}

// ---------------------------------------------------------------------------
// K1: group norm. One block per (batch, group): 48 ch x 4096. Two passes,
// second pass transposes through LDS and stores XT f16 [b][n][c].
// ---------------------------------------------------------------------------
__global__ __launch_bounds__(256) void k_gnorm(
    const float* __restrict__ x, const float* __restrict__ gw,
    const float* __restrict__ gb, _Float16* __restrict__ XT) {
  const int CG = 48, TOT = CG * NDIM;
  int b = blockIdx.x >> 3, g = blockIdx.x & 7;
  const float* xp = x + ((size_t)b * CDIM + g * CG) * NDIM;

  float s = 0.f, ss = 0.f;
  for (int i = threadIdx.x; i < TOT; i += 256) {
    float v = xp[i];
    s += v; ss += v * v;
  }
  for (int off = 1; off < 32; off <<= 1) {
    s  += __shfl_xor(s, off);
    ss += __shfl_xor(ss, off);
  }
  __shared__ float rs[8], rss[8], mus[1], rsg[1];
  int w = threadIdx.x >> 5, lane = threadIdx.x & 31;
  if (lane == 0) { rs[w] = s; rss[w] = ss; }
  __syncthreads();
  if (threadIdx.x == 0) {
    float S = 0.f, SS = 0.f;
    for (int i = 0; i < 8; ++i) { S += rs[i]; SS += rss[i]; }
    float mu  = S / (float)TOT;
    float var = SS / (float)TOT - mu * mu;
    mus[0] = mu;
    rsg[0] = rsqrtf(var + 1e-5f);
  }
  __syncthreads();
  float mu = mus[0], rsig = rsg[0];

  __shared__ _Float16 tile[48][72];
  for (int nt = 0; nt < NDIM / 64; ++nt) {
    int nbase = nt * 64;
    for (int e = 0; e < 12; ++e) {
      int li = threadIdx.x + e * 256;     // 0..3071 over 48x64
      int cl = li >> 6, nl = li & 63;
      int c  = g * CG + cl;
      float v  = xp[(size_t)cl * NDIM + nbase + nl];
      float xn = (v - mu) * rsig * gw[c] + gb[c];
      tile[cl][nl] = (_Float16)xn;
    }
    __syncthreads();
    for (int e = 0; e < 12; ++e) {
      int li = threadIdx.x + e * 256;
      int nl = li / 48, cl = li % 48;
      XT[((size_t)b * NDIM + nbase + nl) * CDIM + g * CG + cl] = tile[cl][nl];
    }
    __syncthreads();
  }
}

// ---------------------------------------------------------------------------
// K2: fused QKVG projection GEMM. M=1920, K=384, N=4096 per batch.
// Block tile 64x256 (8 waves 2x4, each wave 32x64 = 2x4 WMMA fragments).
// Section-specific epilogues (sections are 64-aligned so uniform per block).
// ---------------------------------------------------------------------------
__global__ __launch_bounds__(256) void k_qkvg(
    const _Float16* __restrict__ W, const float* __restrict__ bias,
    const _Float16* __restrict__ XT,
    _Float16* __restrict__ Qt, _Float16* __restrict__ Kb,
    _Float16* __restrict__ Vb, _Float16* __restrict__ Ib,
    float* __restrict__ fpart) {
  const int MT = MROWS / 64, NT = NDIM / 256;   // 30, 16
  __shared__ __align__(16) _Float16 lt[8][64][40];

  int bx = blockIdx.x;
  int b  = bx / (MT * NT);
  int r  = bx % (MT * NT);
  int bm = r / NT, bn = r % NT;
  int w = threadIdx.x >> 5, lane = threadIdx.x & 31;
  int wm = w >> 2, wn = w & 3;
  int m0 = bm * 64 + wm * 32;
  int n0 = bn * 256 + wn * 64;
  const _Float16* Bp = XT + (size_t)b * NDIM * CDIM;

  v8f acc[2][4] = {};
  for (int k0 = 0; k0 < CDIM; k0 += 32) {
    v16h a0 = load_frag(W + (size_t)m0 * CDIM + k0, CDIM, lane);
    v16h a1 = load_frag(W + (size_t)(m0 + 16) * CDIM + k0, CDIM, lane);
#pragma unroll
    for (int fj = 0; fj < 4; ++fj) {
      v16h bf = load_frag(Bp + (size_t)(n0 + fj * 16) * CDIM + k0, CDIM, lane);
      acc[0][fj] = wmma16(a0, bf, acc[0][fj]);
      acc[1][fj] = wmma16(a1, bf, acc[1][fj]);
    }
  }

  int col = lane & 15;
  int mhi = (lane >> 4) << 3;
  int sec = m0 / CDIM;      // 0:q 1:k 2:v 3:f 4:i

  if (sec == 0) {           // q: scale + transpose to Qt[n][c]
    for (int fi = 0; fi < 2; ++fi)
      for (int fj = 0; fj < 4; ++fj)
        for (int i = 0; i < 8; ++i) {
          int ml = fi * 16 + mhi + i, nl = fj * 16 + col;
          float v = (acc[fi][fj][i] + bias[m0 + fi * 16 + mhi + i]) * QSCALE;
          lt[w][nl][ml] = (_Float16)v;
        }
    for (int ni = 0; ni < 2; ++ni) {
      int nl = lane + ni * 32;
      _Float16* dst = Qt + ((size_t)b * NDIM + n0 + nl) * CDIM + m0;
      for (int m8 = 0; m8 < 32; m8 += 8)
        *(v8h*)(dst + m8) = *(const v8h*)&lt[w][nl][m8];
    }
  } else if (sec == 1 || sec == 2) {  // raw k / v, row-major [c][n]
    _Float16* dst = (sec == 1) ? Kb : Vb;
    int base = (sec == 1) ? 384 : 768;
    for (int fi = 0; fi < 2; ++fi)
      for (int fj = 0; fj < 4; ++fj)
        for (int i = 0; i < 8; ++i) {
          int rg = m0 + fi * 16 + mhi + i;
          int n  = n0 + fj * 16 + col;
          dst[((size_t)b * CDIM + rg - base) * NDIM + n] =
              (_Float16)(acc[fi][fj][i] + bias[rg]);
        }
  } else if (sec == 3) {    // f-gate: sigmoid + 16-lane reduce -> partials
    for (int fi = 0; fi < 2; ++fi)
      for (int i = 0; i < 8; ++i) {
        int rg = m0 + fi * 16 + mhi + i;
        float bb = bias[rg];
        float s = sigmoidf_(acc[fi][0][i] + bb) + sigmoidf_(acc[fi][1][i] + bb)
                + sigmoidf_(acc[fi][2][i] + bb) + sigmoidf_(acc[fi][3][i] + bb);
        for (int off = 1; off < 16; off <<= 1) s += __shfl_xor(s, off, 32);
        if (col == 0)
          fpart[((size_t)b * CDIM + rg - 1152) * 64 + bn * 4 + wn] = s;
      }
  } else {                  // i-gate: sigmoid, row-major [c][n]
    for (int fi = 0; fi < 2; ++fi)
      for (int fj = 0; fj < 4; ++fj)
        for (int i = 0; i < 8; ++i) {
          int rg = m0 + fi * 16 + mhi + i;
          int n  = n0 + fj * 16 + col;
          Ib[((size_t)b * CDIM + rg - 1536) * NDIM + n] =
              (_Float16)sigmoidf_(acc[fi][fj][i] + bias[rg]);
        }
  }
}

// K2b: deterministic reduction of f-gate partials (64 per (b,c)).
__global__ __launch_bounds__(256) void k_fsum(const float* __restrict__ fpart,
                                              float* __restrict__ fsum) {
  int idx = blockIdx.x * 256 + threadIdx.x;
  if (idx < BDIM * CDIM) {
    const float* p = fpart + (size_t)idx * 64;
    float s = 0.f;
    for (int j = 0; j < 64; ++j) s += p[j];
    fsum[idx] = s;
  }
}

// ---------------------------------------------------------------------------
// K3: knowledge = (k .* i) @ v^T / N.  M=N=384, K=4096. A fused in VGPRs.
// Block tile 128x128 (waves 4x2, each 32x64).
// ---------------------------------------------------------------------------
__global__ __launch_bounds__(256) void k_knowledge(
    const _Float16* __restrict__ Kb, const _Float16* __restrict__ Ib,
    const _Float16* __restrict__ Vb, float* __restrict__ knw) {
  int bx = blockIdx.x;
  int b  = bx / 9;
  int r  = bx % 9;
  int bm = r / 3, bn = r % 3;
  int w = threadIdx.x >> 5, lane = threadIdx.x & 31;
  int wm = w >> 1, wn = w & 1;
  int m0 = bm * 128 + wm * 32;
  int n0 = bn * 128 + wn * 64;
  const _Float16* Kp = Kb + (size_t)b * CDIM * NDIM;
  const _Float16* Ip = Ib + (size_t)b * CDIM * NDIM;
  const _Float16* Vp = Vb + (size_t)b * CDIM * NDIM;

  v8f acc[2][4] = {};
  for (int k0 = 0; k0 < NDIM; k0 += 32) {
    v16h a0 = load_frag_mul(Kp + (size_t)m0 * NDIM + k0,
                            Ip + (size_t)m0 * NDIM + k0, NDIM, lane);
    v16h a1 = load_frag_mul(Kp + (size_t)(m0 + 16) * NDIM + k0,
                            Ip + (size_t)(m0 + 16) * NDIM + k0, NDIM, lane);
#pragma unroll
    for (int fj = 0; fj < 4; ++fj) {
      v16h bf = load_frag(Vp + (size_t)(n0 + fj * 16) * NDIM + k0, NDIM, lane);
      acc[0][fj] = wmma16(a0, bf, acc[0][fj]);
      acc[1][fj] = wmma16(a1, bf, acc[1][fj]);
    }
  }
  int col = lane & 15, mhi = (lane >> 4) << 3;
  const float inv = 1.0f / (float)NDIM;
  for (int fi = 0; fi < 2; ++fi)
    for (int fj = 0; fj < 4; ++fj)
      for (int i = 0; i < 8; ++i) {
        int c = m0 + fi * 16 + mhi + i;
        int d = n0 + fj * 16 + col;
        knw[((size_t)b * CDIM + c) * CDIM + d] = acc[fi][fj][i] * inv;
      }
}

// ---------------------------------------------------------------------------
// K4: new_memory = forget*prev + knw (f32 output) and Mt = memory^T in f16.
// ---------------------------------------------------------------------------
__global__ __launch_bounds__(256) void k_combine(
    const float* __restrict__ prev, const float* __restrict__ knw,
    const float* __restrict__ fsum, float* __restrict__ outmem,
    _Float16* __restrict__ Mt) {
  __shared__ _Float16 t[32][34];
  int bx = blockIdx.x;
  int b  = bx / 144;
  int r  = bx % 144;
  int c0 = (r / 12) * 32, d0 = (r % 12) * 32;
  const float inv = 1.0f / (float)NDIM;
  for (int e = 0; e < 4; ++e) {
    int li = threadIdx.x + e * 256;
    int cl = li >> 5, dl = li & 31;
    int c = c0 + cl, d = d0 + dl;
    size_t idx = ((size_t)b * CDIM + c) * CDIM + d;
    float nm = fsum[b * CDIM + c] * inv * prev[idx] + knw[idx];
    outmem[idx] = nm;
    t[cl][dl] = (_Float16)nm;
  }
  __syncthreads();
  for (int e = 0; e < 4; ++e) {
    int li = threadIdx.x + e * 256;
    int dl = li >> 5, cl = li & 31;
    Mt[((size_t)b * CDIM + d0 + dl) * CDIM + c0 + cl] = t[cl][dl];
  }
}

// ---------------------------------------------------------------------------
// K5: retrieved = M^T q. A=Mt[d][c], B=Qt[n][c]. M=384, N=4096, K=384.
// Epilogue transposes to Rt[n][d] (Rt aliases the dead XT buffer).
// ---------------------------------------------------------------------------
__global__ __launch_bounds__(256) void k_retrieved(
    const _Float16* __restrict__ Mt, const _Float16* __restrict__ Qt,
    _Float16* __restrict__ Rt) {
  const int MT = CDIM / 64, NT = NDIM / 256;   // 6, 16
  __shared__ __align__(16) _Float16 lt[8][64][40];
  int bx = blockIdx.x;
  int b  = bx / (MT * NT);
  int r  = bx % (MT * NT);
  int bm = r / NT, bn = r % NT;
  int w = threadIdx.x >> 5, lane = threadIdx.x & 31;
  int wm = w >> 2, wn = w & 3;
  int m0 = bm * 64 + wm * 32;
  int n0 = bn * 256 + wn * 64;
  const _Float16* Ap = Mt + (size_t)b * CDIM * CDIM;
  const _Float16* Bp = Qt + (size_t)b * NDIM * CDIM;

  v8f acc[2][4] = {};
  for (int k0 = 0; k0 < CDIM; k0 += 32) {
    v16h a0 = load_frag(Ap + (size_t)m0 * CDIM + k0, CDIM, lane);
    v16h a1 = load_frag(Ap + (size_t)(m0 + 16) * CDIM + k0, CDIM, lane);
#pragma unroll
    for (int fj = 0; fj < 4; ++fj) {
      v16h bf = load_frag(Bp + (size_t)(n0 + fj * 16) * CDIM + k0, CDIM, lane);
      acc[0][fj] = wmma16(a0, bf, acc[0][fj]);
      acc[1][fj] = wmma16(a1, bf, acc[1][fj]);
    }
  }
  int col = lane & 15, mhi = (lane >> 4) << 3;
  for (int fi = 0; fi < 2; ++fi)
    for (int fj = 0; fj < 4; ++fj)
      for (int i = 0; i < 8; ++i)
        lt[w][fj * 16 + col][fi * 16 + mhi + i] = (_Float16)acc[fi][fj][i];
  for (int ni = 0; ni < 2; ++ni) {
    int nl = lane + ni * 32;
    _Float16* dst = Rt + ((size_t)b * NDIM + n0 + nl) * CDIM + m0;
    for (int m8 = 0; m8 < 32; m8 += 8)
      *(v8h*)(dst + m8) = *(const v8h*)&lt[w][nl][m8];
  }
}

// ---------------------------------------------------------------------------
// K6: out = Wp @ retrieved + bp + x (f32 output).
// ---------------------------------------------------------------------------
__global__ __launch_bounds__(256) void k_out(
    const _Float16* __restrict__ Wp16, const float* __restrict__ bp,
    const _Float16* __restrict__ Rt, const float* __restrict__ x,
    float* __restrict__ out) {
  const int MT = CDIM / 64, NT = NDIM / 256;
  int bx = blockIdx.x;
  int b  = bx / (MT * NT);
  int r  = bx % (MT * NT);
  int bm = r / NT, bn = r % NT;
  int w = threadIdx.x >> 5, lane = threadIdx.x & 31;
  int wm = w >> 2, wn = w & 3;
  int m0 = bm * 64 + wm * 32;
  int n0 = bn * 256 + wn * 64;
  const _Float16* Bp = Rt + (size_t)b * NDIM * CDIM;

  v8f acc[2][4] = {};
  for (int k0 = 0; k0 < CDIM; k0 += 32) {
    v16h a0 = load_frag(Wp16 + (size_t)m0 * CDIM + k0, CDIM, lane);
    v16h a1 = load_frag(Wp16 + (size_t)(m0 + 16) * CDIM + k0, CDIM, lane);
#pragma unroll
    for (int fj = 0; fj < 4; ++fj) {
      v16h bf = load_frag(Bp + (size_t)(n0 + fj * 16) * CDIM + k0, CDIM, lane);
      acc[0][fj] = wmma16(a0, bf, acc[0][fj]);
      acc[1][fj] = wmma16(a1, bf, acc[1][fj]);
    }
  }
  int col = lane & 15, mhi = (lane >> 4) << 3;
  for (int fi = 0; fi < 2; ++fi)
    for (int fj = 0; fj < 4; ++fj)
      for (int i = 0; i < 8; ++i) {
        int o = m0 + fi * 16 + mhi + i;
        int n = n0 + fj * 16 + col;
        size_t idx = ((size_t)b * CDIM + o) * NDIM + n;
        out[idx] = acc[fi][fj][i] + bp[o] + x[idx];
      }
}

// ---------------------------------------------------------------------------
extern "C" void kernel_launch(void* const* d_in, const int* in_sizes, int n_in,
                              void* d_out, int out_size, void* d_ws,
                              size_t ws_size, hipStream_t stream) {
  (void)in_sizes; (void)n_in; (void)out_size; (void)ws_size;
  const float* x    = (const float*)d_in[0];
  const float* prev = (const float*)d_in[1];
  const float* nw   = (const float*)d_in[2];
  const float* nb   = (const float*)d_in[3];
  const float* Wq   = (const float*)d_in[4];
  const float* bq   = (const float*)d_in[5];
  const float* Wk   = (const float*)d_in[6];
  const float* bk   = (const float*)d_in[7];
  const float* Wv   = (const float*)d_in[8];
  const float* bv   = (const float*)d_in[9];
  const float* Wg   = (const float*)d_in[10];
  const float* bg   = (const float*)d_in[11];
  const float* Wp   = (const float*)d_in[12];
  const float* bp   = (const float*)d_in[13];

  float* out    = (float*)d_out;
  float* outmem = out + (size_t)BDIM * CDIM * NDIM;   // second tuple element

  char* ws = (char*)d_ws;
  const size_t actB = (size_t)BDIM * NDIM * CDIM * sizeof(_Float16); // 48 MiB
  size_t oXT = 0;                       // XT f16 [b][n][c]; later aliased by Rt
  size_t oQ  = oXT + actB;              // Qt f16 [b][n][c]
  size_t oK  = oQ + actB;               // Kb f16 [b][c][n]
  size_t oV  = oK + actB;               // Vb f16 [b][c][n]
  size_t oI  = oV + actB;               // Ib f16 [b][c][n]
  size_t oW  = oI + actB;               // Wc16 [1920][384]
  size_t oWp = oW + (size_t)MROWS * CDIM * 2;
  size_t oB  = oWp + (size_t)CDIM * CDIM * 2;
  size_t oFp = oB + MROWS * 4;          // f-gate partials [b][c][64]
  size_t oFs = oFp + (size_t)BDIM * CDIM * 64 * 4;
  size_t oKn = oFs + (size_t)BDIM * CDIM * 4;   // knowledge f32 [b][c][c]
  size_t oMt = oKn + (size_t)BDIM * CDIM * CDIM * 4;

  _Float16* XT   = (_Float16*)(ws + oXT);
  _Float16* Qt   = (_Float16*)(ws + oQ);
  _Float16* Kb   = (_Float16*)(ws + oK);
  _Float16* Vb   = (_Float16*)(ws + oV);
  _Float16* Ib   = (_Float16*)(ws + oI);
  _Float16* Wc16 = (_Float16*)(ws + oW);
  _Float16* Wp16 = (_Float16*)(ws + oWp);
  float*    bcat = (float*)(ws + oB);
  float*    fpart= (float*)(ws + oFp);
  float*    fsum = (float*)(ws + oFs);
  float*    knw  = (float*)(ws + oKn);
  _Float16* Mt   = (_Float16*)(ws + oMt);
  _Float16* Rt   = XT;                  // safe alias: XT dead after k_qkvg

  k_prep<<<(MROWS * CDIM + 255) / 256, 256, 0, stream>>>(
      Wq, Wk, Wv, Wg, Wp, bq, bk, bv, bg, Wc16, bcat, Wp16);
  k_gnorm<<<BDIM * 8, 256, 0, stream>>>(x, nw, nb, XT);
  k_qkvg<<<BDIM * (MROWS / 64) * (NDIM / 256), 256, 0, stream>>>(
      Wc16, bcat, XT, Qt, Kb, Vb, Ib, fpart);
  k_fsum<<<(BDIM * CDIM + 255) / 256, 256, 0, stream>>>(fpart, fsum);
  k_knowledge<<<BDIM * 3 * 3, 256, 0, stream>>>(Kb, Ib, Vb, knw);
  k_combine<<<BDIM * 12 * 12, 256, 0, stream>>>(prev, knw, fsum, outmem, Mt);
  k_retrieved<<<BDIM * (CDIM / 64) * (NDIM / 256), 256, 0, stream>>>(Mt, Qt, Rt);
  k_out<<<BDIM * (CDIM / 64) * (NDIM / 256), 256, 0, stream>>>(
      Wp16, bp, Rt, x, out);
}
